// VFNet_60756607369839
// MI455X (gfx1250) — matrix operand order repaired and checked by reading, not compiled, not run on hardware
//
#include <hip/hip_runtime.h>
#include <hip/hip_bf16.h>

// ---------------------------------------------------------------------------
// VFNet forward on gfx1250 (MI455X). Heavy convs run as f16 WMMA implicit
// GEMM (v_wmma_f32_16x16x32_f16, f32 accum); gather stages are VALU kernels.
// Weight staging uses GLOBAL_LOAD_ASYNC_TO_LDS_B128 when the toolchain
// exposes it (ASYNCcnt path), else a global_load_b128 + ds_store_b128 copy.
// ---------------------------------------------------------------------------

typedef __attribute__((ext_vector_type(16))) _Float16 v16h;
typedef __attribute__((ext_vector_type(8)))  float    v8f;

#if __has_builtin(__builtin_amdgcn_global_load_async_to_lds_b128)
#define ASYNC_LDS 1
#endif

#ifdef ASYNC_LDS
// Builtin signature (from clang diagnostic): param0 is int4* in AS(1)
// (printed as "__vector_size__(4*sizeof(int)) int __device__ *"),
// param1 the LDS-side int4* in AS(3); then imm offset, imm cpol.
typedef int v4i_vec __attribute__((vector_size(4 * sizeof(int))));
typedef __attribute__((address_space(1))) v4i_vec* gv4p_t;
typedef __attribute__((address_space(3))) v4i_vec* lv4p_t;
__device__ __forceinline__ void async_a_copy_b128(const _Float16* gsrc, _Float16* ldst) {
    __builtin_amdgcn_global_load_async_to_lds_b128(
        (gv4p_t)(gsrc), (lv4p_t)(ldst), 0, 0);
}
__device__ __forceinline__ void async_wait0() {
#if __has_builtin(__builtin_amdgcn_s_wait_asynccnt)
    __builtin_amdgcn_s_wait_asynccnt(0);
#else
    asm volatile("s_wait_asynccnt 0" ::: "memory");
#endif
}
#endif

#define NCAM 6
#define CIN0 64
#define HF 44
#define WF 80
#define PIX (HF * WF)            // 3520
#define XD 100
#define YD 100
#define ZD 20
#define NVOX (XD * YD * ZD)      // 200000
#define DBN 50
#define VDIM 64
#define FOUT 256
#define CH1 (VDIM * DBN)         // 3200
#define K3D (128 * 27)           // 3456
#define K_R1 (CH1 * 9)           // 28800
#define K_R2 (FOUT * 9)          // 2304
#define EPSV 1e-8f

// nonuniform x axis: ex[i] = 1 / (1/X0 - i*EXSTEP), ascending 2.0 .. 51.5
#define EXB   0.5f
#define EXA   (1.0f / 51.5f)
#define EXSTEP ((EXB - EXA) / 99.0f)
// inverse-depth bins: dbin[i] = 1 / (1/2 - i*DSTEP), ascending 2 .. 50
#define DBB   0.5f
#define DBA   (1.0f / 50.0f)
#define DSTEP ((DBB - DBA) / 49.0f)

__device__ __forceinline__ float eluf(float x) {
    return x > 0.0f ? x : (expf(x) - 1.0f);
}
__device__ __forceinline__ float ex_of(int i) {
    return 1.0f / (EXB - (float)i * EXSTEP);
}
__device__ __forceinline__ float dbin_of(int i) {
    return 1.0f / (DBB - (float)i * DSTEP);
}
// metric x -> continuous index on the nonuniform axis (searchsorted + lerp)
__device__ __forceinline__ float nonuni_x(float q) {
    float qc = fminf(fmaxf(q, 2.0f), 51.5f - 1e-6f);
    float fi = (EXB - 1.0f / qc) / EXSTEP;      // analytic inverse of ex[]
    int k = (int)floorf(fi);
    k = k < 0 ? 0 : (k > 98 ? 98 : k);
    float exk  = ex_of(k);
    float exk1 = ex_of(k + 1);
    float t = (qc - exk) / (exk1 - exk + 1e-12f);
    return (float)k + t;
}

// ---------------------------------------------------------------------------
// WMMA fragment loaders from LDS, per CDNA5 ISA 7.12.2 layouts (wave32).
// A: 16x32 f16, lanes 0-15 hold K{0..7,16..23}, lanes 16-31 hold K{8..15,24..31}
// B: 32x16 f16, col N=lane%16, lanes 0-15 hold K{0..7,16..23}, lanes 16-31 hold K{8..15,24..31}
// ---------------------------------------------------------------------------
__device__ __forceinline__ v16h load_a_frag(const _Float16* As, int m0, int lane) {
    int l = lane & 15, hi = (lane >> 4) & 1;
    const _Float16* r = As + (m0 + l) * 32 + hi * 8;
    v16h a;
#pragma unroll
    for (int i = 0; i < 8; ++i) { a[i] = r[i]; a[i + 8] = r[i + 16]; }
    return a;
}
__device__ __forceinline__ v16h load_b_frag(const _Float16* Bs, int lane) {
    int l = lane & 15, hi = (lane >> 4) & 1;
    const _Float16* r = Bs + l * 32 + hi * 8;
    v16h b;
#pragma unroll
    for (int i = 0; i < 8; ++i) { b[i] = r[i]; b[i + 8] = r[i + 16]; }
    return b;
}

// ---------------------------------------------------------------------------
// Kernel 1: f32 -> f16 weight repack (flat layouts already match GEMM K order)
// ---------------------------------------------------------------------------
__global__ void f32tof16_kernel(const float* __restrict__ src,
                                _Float16* __restrict__ dst, int n) {
    int i = blockIdx.x * blockDim.x + threadIdx.x;
    if (i < n) dst[i] = (_Float16)src[i];
}

// ---------------------------------------------------------------------------
// Kernel 2: backproject surround features into the voxel grid + ov/no head.
// One block (96 thr) per voxel. Writes vf[64][NVOX] f32.
// ---------------------------------------------------------------------------
__global__ __launch_bounds__(96) void backproject_kernel(
    const float* __restrict__ feats, const float* __restrict__ masks,
    const float* __restrict__ Kc,    const float* __restrict__ ext,
    const float* __restrict__ w_ov,  const float* __restrict__ b_ov,
    const float* __restrict__ w_no,  const float* __restrict__ b_no,
    float* __restrict__ vf) {
    __shared__ float su[NCAM], sv[NCAM];
    __shared__ int   sval[NCAM];
    __shared__ float fr[65], bk[65];
    __shared__ int   scnt;

    int n = blockIdx.x;
    int x = n % XD, y = (n / XD) % YD, z = n / (XD * YD);
    int t = threadIdx.x;

    if (t < NCAM) {
        int c = t;
        float vx = ex_of(x);
        float vy = -25.0f + (float)y * (49.5f / 99.0f);
        float vz = -3.0f  + (float)z * (5.7f  / 19.0f);
        const float* E = ext + c * 16;
        // rigid inverse: pc = R^T (vp - t)
        float dx = vx - E[3], dy = vy - E[7], dz = vz - E[11];
        float p0 = E[0] * dx + E[4] * dy + E[8]  * dz;
        float p1 = E[1] * dx + E[5] * dy + E[9]  * dz;
        float p2 = E[2] * dx + E[6] * dy + E[10] * dz;
        const float* Km = Kc + c * 16;
        float px = Km[0]  * p0 + Km[1]  * p1 + Km[2]  * p2 + Km[3];
        float py = Km[4]  * p0 + Km[5]  * p1 + Km[6]  * p2 + Km[7];
        float pz = Km[8]  * p0 + Km[9]  * p1 + Km[10] * p2 + Km[11];
        float u = px / (pz + EPSV), v = py / (pz + EPSV);
        int valid = (pz > EPSV) && (u >= 0.0f) && (u <= (float)(WF - 1)) &&
                    (v >= 0.0f) && (v <= (float)(HF - 1));
        su[c] = u; sv[c] = v; sval[c] = valid;
    }
    __syncthreads();

    if (t < 65) {
        float f = 0.0f, b = 0.0f;
        for (int c = 0; c < NCAM; ++c) {
            if (!sval[c]) continue;
            const float* img = (t < CIN0) ? feats + ((size_t)(c * CIN0 + t)) * PIX
                                          : masks + (size_t)c * PIX;
            float u = su[c], v = sv[c];
            float u0 = floorf(u), v0 = floorf(v);
            float du = u - u0, dv = v - v0;
            int u0i = (int)u0; u0i = u0i < 0 ? 0 : (u0i > WF - 1 ? WF - 1 : u0i);
            int v0i = (int)v0; v0i = v0i < 0 ? 0 : (v0i > HF - 1 ? HF - 1 : v0i);
            int u1i = u0i + 1 > WF - 1 ? WF - 1 : u0i + 1;
            int v1i = v0i + 1 > HF - 1 ? HF - 1 : v0i + 1;
            float s = img[v0i * WF + u0i] * (1.0f - du) * (1.0f - dv)
                    + img[v0i * WF + u1i] * du * (1.0f - dv)
                    + img[v1i * WF + u0i] * (1.0f - du) * dv
                    + img[v1i * WF + u1i] * du * dv;
            if (c < 3) f += s; else b += s;
        }
        fr[t] = f; bk[t] = b;
    }
    if (t == 0) {
        int c = 0;
        for (int i = 0; i < NCAM; ++i) c += sval[i];
        scnt = c;
    }
    __syncthreads();

    if (t < VDIM) {
        int cnt = scnt;
        float o = 0.0f;
        if (cnt == 2) {
            const float* wr = w_ov + t * 130;
            float s = b_ov[t];
            for (int j = 0; j < 65; ++j) s += wr[j] * fr[j] + wr[65 + j] * bk[j];
            o = eluf(s);
        } else if (cnt == 1) {
            const float* wr = w_no + t * 65;
            float s = b_no[t];
            for (int j = 0; j < 65; ++j) s += wr[j] * (fr[j] + bk[j]);
            o = eluf(s);
        }
        vf[(size_t)t * NVOX + n] = o;
    }
}

// ---------------------------------------------------------------------------
// Kernel 3: nonuniform-x -> uniform-x remap (1-D lerp in x; z,y are integral)
// and f16 pack of the 128-channel conv3d input.
// ---------------------------------------------------------------------------
__global__ void remap_kernel(const float* __restrict__ vf,
                             _Float16* __restrict__ cin) {
    int i = blockIdx.x * blockDim.x + threadIdx.x;
    if (i >= VDIM * NVOX) return;
    int ch = i / NVOX, n = i % NVOX;
    int x = n % XD;
    int rowbase = n - x;                       // (z*YD + y)*XD
    cin[i] = (_Float16)vf[i];
    float q  = 2.0f + (float)x * (49.5f / 99.0f);   // uniform metric x
    float fi = nonuni_x(q);
    int k = (int)floorf(fi);
    k = k < 0 ? 0 : (k > 98 ? 98 : k);
    float tt = fi - (float)k;
    const float* base = vf + (size_t)ch * NVOX + rowbase;
    float rm = base[k] * (1.0f - tt) + base[k + 1] * tt;
    cin[(size_t)(VDIM + ch) * NVOX + n] = (_Float16)rm;
}

// ---------------------------------------------------------------------------
// Kernel 4: conv3d 3x3x3 fusion as WMMA implicit GEMM.
// M=64 (4 waves x one 16-tile), N=16 voxels per block, K=3456 in 32-chunks.
// (ch, r) for the im2col decode is carried incrementally (k += 32 = 27 + 5)
// to keep the VALU pipe light next to the WMMA pipe.
// ---------------------------------------------------------------------------
__global__ __launch_bounds__(128) void conv3d_wmma_kernel(
    const _Float16* __restrict__ in,  // [128][NVOX]
    const _Float16* __restrict__ wt,  // [64][3456]
    const float* __restrict__ bias,   // [64]
    float* __restrict__ out) {        // [64][NVOX]
    __shared__ _Float16 As[64 * 32];
    __shared__ _Float16 Bs[16 * 32];
    int t = threadIdx.x, lane = t & 31, wave = t >> 5;
    int n0 = blockIdx.x * 16;
    int bbase = t * 4, bn = bbase >> 5, bko = bbase & 31;
    int nB = n0 + bn;
    int xB = nB % XD, yB = (nB / XD) % YD, zB = nB / (XD * YD);
    int chj[4], rj[4];
#pragma unroll
    for (int j = 0; j < 4; ++j) {
        int k = bko + j;
        chj[j] = k / 27; rj[j] = k % 27;
    }
    v8f acc = {};
    for (int kc = 0; kc < K3D; kc += 32) {
        {   // A: 2048 halves, 16 per thread (two b128 transfers)
            const _Float16* src = wt + (size_t)(t >> 1) * K3D + kc + (t & 1) * 16;
            _Float16* dst = As + (t >> 1) * 32 + (t & 1) * 16;
#ifdef ASYNC_LDS
            async_a_copy_b128(src, dst);
            async_a_copy_b128(src + 8, dst + 8);
#else
            const uint4* s4 = (const uint4*)src;
            uint4* d4 = (uint4*)dst;
            d4[0] = s4[0]; d4[1] = s4[1];
#endif
            __builtin_prefetch((const void*)(src + 32), 0, 0);  // next K chunk
        }
#pragma unroll
        for (int j = 0; j < 4; ++j) {   // B: im2col gather, 4 halves/thread
            int r = rj[j];
            int kd = r / 9, kh = (r % 9) / 3, kw = r % 3;
            int zz = zB + kd - 1, yy = yB + kh - 1, xx = xB + kw - 1;
            _Float16 v = (_Float16)0.0f;
            if (zz >= 0 && zz < ZD && yy >= 0 && yy < YD && xx >= 0 && xx < XD)
                v = in[(size_t)chj[j] * NVOX + (zz * YD + yy) * XD + xx];
            Bs[bn * 32 + bko + j] = v;
            int rn = r + 5, cn = chj[j] + 1;        // advance k by 32 = 27 + 5
            if (rn >= 27) { rn -= 27; cn += 1; }
            rj[j] = rn; chj[j] = cn;
        }
#ifdef ASYNC_LDS
        async_wait0();
#endif
        __syncthreads();
        v16h bf = load_b_frag(Bs, lane);
        v16h af = load_a_frag(As, wave * 16, lane);
        acc = __builtin_amdgcn_wmma_f32_16x16x32_f16(
            false, af, false, bf, (short)0, acc, false, false);
        __syncthreads();
    }
    int l = lane & 15, hi = lane >> 4;
    int n = n0 + l;
#pragma unroll
    for (int r = 0; r < 8; ++r) {       // f32 C/D layout: M = r + hi*8
        int m = wave * 16 + r + hi * 8;
        out[(size_t)m * NVOX + n] = eluf(acc[r] + bias[m]);
    }
}

// ---------------------------------------------------------------------------
// Kernel 5: project fused voxels into camera frusta (trilinear over vol),
// f16 pack with channel order vd*DB + d.
// ---------------------------------------------------------------------------
__global__ __launch_bounds__(256) void project_kernel(
    const float* __restrict__ Kc, const float* __restrict__ ext,
    const float* __restrict__ vol, _Float16* __restrict__ proj) {
    int tid = blockIdx.x * blockDim.x + threadIdx.x;
    if (tid >= NCAM * DBN * PIX) return;
    int p = tid % PIX, d = (tid / PIX) % DBN, cam = tid / (PIX * DBN);
    float u = (float)(p % WF), v = (float)(p / WF);

    const float* Km = Kc + cam * 16;    // 3x3 inverse via adjugate
    float k00 = Km[0], k01 = Km[1], k02 = Km[2];
    float k10 = Km[4], k11 = Km[5], k12 = Km[6];
    float k20 = Km[8], k21 = Km[9], k22 = Km[10];
    float det = k00 * (k11 * k22 - k12 * k21) - k01 * (k10 * k22 - k12 * k20)
              + k02 * (k10 * k21 - k11 * k20);
    float id = 1.0f / det;
    float dx = ((k11 * k22 - k12 * k21) * u + (k02 * k21 - k01 * k22) * v + (k01 * k12 - k02 * k11)) * id;
    float dy = ((k12 * k20 - k10 * k22) * u + (k00 * k22 - k02 * k20) * v + (k02 * k10 - k00 * k12)) * id;
    float dz = ((k10 * k21 - k11 * k20) * u + (k01 * k20 - k00 * k21) * v + (k00 * k11 - k01 * k10)) * id;

    float dep = dbin_of(d);
    float px = dx * dep, py = dy * dep, pz = dz * dep;
    const float* E = ext + cam * 16;
    float wx = E[0] * px + E[1] * py + E[2]  * pz + E[3];
    float wy = E[4] * px + E[5] * py + E[6]  * pz + E[7];
    float wz = E[8] * px + E[9] * py + E[10] * pz + E[11];

    float xi = nonuni_x(wx);
    float yi = (wy + 25.0f) / 49.5f * 99.0f;
    float zi = (wz + 3.0f)  / 5.7f  * 19.0f;
    bool inb = (wx >= 2.0f) && (wx <= 51.5f) && (yi >= 0.0f) && (yi <= 99.0f)
            && (zi >= 0.0f) && (zi <= 19.0f);

    _Float16* dst = proj + (size_t)cam * CH1 * PIX + (size_t)d * PIX + p;
    if (!inb) {
        for (int vd = 0; vd < VDIM; ++vd) dst[(size_t)vd * DBN * PIX] = (_Float16)0.0f;
        return;
    }
    float fx = floorf(xi), fy = floorf(yi), fz = floorf(zi);
    float tx = xi - fx, ty = yi - fy, tz = zi - fz;
    int x0 = (int)fx; x0 = x0 < 0 ? 0 : (x0 > XD - 1 ? XD - 1 : x0);
    int y0 = (int)fy; y0 = y0 < 0 ? 0 : (y0 > YD - 1 ? YD - 1 : y0);
    int z0 = (int)fz; z0 = z0 < 0 ? 0 : (z0 > ZD - 1 ? ZD - 1 : z0);
    int x1 = x0 + 1 > XD - 1 ? XD - 1 : x0 + 1;
    int y1 = y0 + 1 > YD - 1 ? YD - 1 : y0 + 1;
    int z1 = z0 + 1 > ZD - 1 ? ZD - 1 : z0 + 1;
    int o000 = (z0 * YD + y0) * XD + x0, o001 = (z0 * YD + y0) * XD + x1;
    int o010 = (z0 * YD + y1) * XD + x0, o011 = (z0 * YD + y1) * XD + x1;
    int o100 = (z1 * YD + y0) * XD + x0, o101 = (z1 * YD + y0) * XD + x1;
    int o110 = (z1 * YD + y1) * XD + x0, o111 = (z1 * YD + y1) * XD + x1;
    float w000 = (1 - tx) * (1 - ty) * (1 - tz), w001 = tx * (1 - ty) * (1 - tz);
    float w010 = (1 - tx) * ty * (1 - tz),       w011 = tx * ty * (1 - tz);
    float w100 = (1 - tx) * (1 - ty) * tz,       w101 = tx * (1 - ty) * tz;
    float w110 = (1 - tx) * ty * tz,             w111 = tx * ty * tz;
    for (int vd = 0; vd < VDIM; ++vd) {
        const float* b = vol + (size_t)vd * NVOX;
        float s = w000 * b[o000] + w001 * b[o001] + w010 * b[o010] + w011 * b[o011]
                + w100 * b[o100] + w101 * b[o101] + w110 * b[o110] + w111 * b[o111];
        dst[(size_t)vd * DBN * PIX] = (_Float16)s;
    }
}

// ---------------------------------------------------------------------------
// Kernels 6/7: 3x3 conv2d as WMMA implicit GEMM, M=256 (8 waves x 2 tiles),
// N=16 pixels per block, K=CIN*9 in 32-chunks through LDS.
// im2col (ch, r) decode carried incrementally (k += 32 = 3*9 + 5).
// ---------------------------------------------------------------------------
template <int CIN, bool OUTF32>
__global__ __launch_bounds__(256) void conv2d_wmma_kernel(
    const _Float16* __restrict__ in,  // [NCAM][CIN][HF][WF]
    const _Float16* __restrict__ wt,  // [256][CIN*9]
    const float* __restrict__ bias,   // [256]
    float* __restrict__ outf,         // used if OUTF32
    _Float16* __restrict__ outh) {    // used if !OUTF32
    constexpr int K = CIN * 9;
    __shared__ _Float16 As[256 * 32];
    __shared__ _Float16 Bs[16 * 32];
    int t = threadIdx.x, lane = t & 31, wave = t >> 5;
    int cam = blockIdx.y;
    int p0 = blockIdx.x * 16;
    const _Float16* inC = in + (size_t)cam * CIN * PIX;
    int bbase = t * 2, bn = bbase >> 5, bko = bbase & 31;
    int pB = p0 + bn, hB = pB / WF, wB = pB % WF;
    int chj[2], rj[2];
#pragma unroll
    for (int j = 0; j < 2; ++j) {
        int k = bko + j;
        chj[j] = k / 9; rj[j] = k % 9;
    }
    v8f acc0 = {}, acc1 = {};
    for (int kc = 0; kc < K; kc += 32) {
        {   // A: 8192 halves, row t, 32 halves (four b128 transfers)
            const _Float16* src = wt + (size_t)t * K + kc;
            _Float16* dst = As + t * 32;
#ifdef ASYNC_LDS
#pragma unroll
            for (int i = 0; i < 4; ++i)
                async_a_copy_b128(src + i * 8, dst + i * 8);
#else
            const uint4* s4 = (const uint4*)src;
            uint4* d4 = (uint4*)dst;
            d4[0] = s4[0]; d4[1] = s4[1]; d4[2] = s4[2]; d4[3] = s4[3];
#endif
            __builtin_prefetch((const void*)(src + 32), 0, 0);  // next K chunk
        }
#pragma unroll
        for (int j = 0; j < 2; ++j) {   // B: im2col gather, 2 halves/thread
            int r = rj[j];
            int kh = r / 3, kw = r % 3;
            int hh = hB + kh - 1, ww = wB + kw - 1;
            _Float16 v = (_Float16)0.0f;
            if (hh >= 0 && hh < HF && ww >= 0 && ww < WF)
                v = inC[(size_t)chj[j] * PIX + hh * WF + ww];
            Bs[bn * 32 + bko + j] = v;
            int rn = r + 5, cn = chj[j] + 3;        // advance k by 32 = 3*9 + 5
            if (rn >= 9) { rn -= 9; cn += 1; }
            rj[j] = rn; chj[j] = cn;
        }
#ifdef ASYNC_LDS
        async_wait0();
#endif
        __syncthreads();
        v16h bf = load_b_frag(Bs, lane);
        v16h a0 = load_a_frag(As, wave * 32, lane);
        v16h a1 = load_a_frag(As, wave * 32 + 16, lane);
        acc0 = __builtin_amdgcn_wmma_f32_16x16x32_f16(
            false, a0, false, bf, (short)0, acc0, false, false);
        acc1 = __builtin_amdgcn_wmma_f32_16x16x32_f16(
            false, a1, false, bf, (short)0, acc1, false, false);
        __syncthreads();
    }
    int l = lane & 15, hi = lane >> 4;
    int n = p0 + l;
#pragma unroll
    for (int r = 0; r < 8; ++r) {
        int m0 = wave * 32 + r + hi * 8;
        int m1 = m0 + 16;
        float x0 = eluf(acc0[r] + bias[m0]);
        float x1 = eluf(acc1[r] + bias[m1]);
        if (OUTF32) {
            outf[((size_t)cam * FOUT + m0) * PIX + n] = x0;
            outf[((size_t)cam * FOUT + m1) * PIX + n] = x1;
        } else {
            outh[((size_t)cam * FOUT + m0) * PIX + n] = (_Float16)x0;
            outh[((size_t)cam * FOUT + m1) * PIX + n] = (_Float16)x1;
        }
    }
}

// ---------------------------------------------------------------------------
// Workspace layout (bytes, all 256-aligned). vf region is reused for h.
// ---------------------------------------------------------------------------
#define OFF_VF   ((size_t)0)            // f32 [64][NVOX]  (later: h f16)
#define OFF_CIN  ((size_t)51200000)     // f16 [128][NVOX]
#define OFF_VOL  ((size_t)102400000)    // f32 [64][NVOX]
#define OFF_PROJ ((size_t)153600000)    // f16 [6][3200][44][80]
#define OFF_WFU  ((size_t)288768000)    // f16 [64][3456]
#define OFF_WR1  ((size_t)289210368)    // f16 [256][28800]
#define OFF_WR2  ((size_t)303955968)    // f16 [256][2304]

extern "C" void kernel_launch(void* const* d_in, const int* in_sizes, int n_in,
                              void* d_out, int out_size, void* d_ws, size_t ws_size,
                              hipStream_t stream) {
    (void)in_sizes; (void)n_in; (void)out_size; (void)ws_size;
    const float* feats = (const float*)d_in[0];
    const float* masks = (const float*)d_in[1];
    const float* Kc    = (const float*)d_in[2];
    const float* ext   = (const float*)d_in[3];
    const float* w_ov  = (const float*)d_in[4];
    const float* b_ov  = (const float*)d_in[5];
    const float* w_no  = (const float*)d_in[6];
    const float* b_no  = (const float*)d_in[7];
    const float* w_fu  = (const float*)d_in[8];
    const float* b_fu  = (const float*)d_in[9];
    const float* w_r1  = (const float*)d_in[10];
    const float* b_r1  = (const float*)d_in[11];
    const float* w_r2  = (const float*)d_in[12];
    const float* b_r2  = (const float*)d_in[13];
    float* out = (float*)d_out;

    char* ws = (char*)d_ws;
    float*    vf   = (float*)(ws + OFF_VF);
    _Float16* cin  = (_Float16*)(ws + OFF_CIN);
    float*    vol  = (float*)(ws + OFF_VOL);
    _Float16* proj = (_Float16*)(ws + OFF_PROJ);
    _Float16* hbuf = (_Float16*)(ws + OFF_VF);     // reuse vf region
    _Float16* wfu16 = (_Float16*)(ws + OFF_WFU);
    _Float16* wr116 = (_Float16*)(ws + OFF_WR1);
    _Float16* wr216 = (_Float16*)(ws + OFF_WR2);

    // 1) weight repack f32 -> f16
    {
        int n1 = VDIM * K3D;            // 221184
        int n2 = FOUT * K_R1;           // 7372800
        int n3 = FOUT * K_R2;           // 589824
        f32tof16_kernel<<<(n1 + 255) / 256, 256, 0, stream>>>(w_fu, wfu16, n1);
        f32tof16_kernel<<<(n2 + 255) / 256, 256, 0, stream>>>(w_r1, wr116, n2);
        f32tof16_kernel<<<(n3 + 255) / 256, 256, 0, stream>>>(w_r2, wr216, n3);
    }
    // 2) backproject + ov/no head -> vf
    backproject_kernel<<<NVOX, 96, 0, stream>>>(feats, masks, Kc, ext,
                                                w_ov, b_ov, w_no, b_no, vf);
    // 3) nonuniform-x remap + f16 pack -> cin [128][NVOX]
    {
        int n = VDIM * NVOX;            // 12.8M
        remap_kernel<<<(n + 255) / 256, 256, 0, stream>>>(vf, cin);
    }
    // 4) conv3d fusion (WMMA) -> vol [64][NVOX]
    conv3d_wmma_kernel<<<NVOX / 16, 128, 0, stream>>>(cin, wfu16, b_fu, vol);
    // 5) frustum projection -> proj f16 [6][3200][44][80]
    {
        int n = NCAM * DBN * PIX;       // 1.056M
        project_kernel<<<(n + 255) / 256, 256, 0, stream>>>(Kc, ext, vol, proj);
    }
    // 6) conv_r1 (WMMA) -> h f16 [6][256][44][80]
    conv2d_wmma_kernel<CH1, false><<<dim3(PIX / 16, NCAM), 256, 0, stream>>>(
        proj, wr116, b_r1, nullptr, hbuf);
    // 7) conv_r2 (WMMA) -> out f32 [6][256][44][80]
    conv2d_wmma_kernel<FOUT, true><<<dim3(PIX / 16, NCAM), 256, 0, stream>>>(
        hbuf, wr216, b_r2, out, nullptr);
}
